// LSTM_1975684956950
// MI455X (gfx1250) — matrix operand order repaired and checked by reading, not compiled, not run on hardware
//
#include <hip/hip_runtime.h>
#include <hip/hip_bf16.h>

#define T_STEPS 512
#define BATCH   64
#define EDIM    512
#define HDIM    1024
#define KF      1536   // E + H
#define NG      4096   // 4 * H
#define XCH     (EDIM / 32)   // 16 x-chunks
#define ALLCH   (KF / 32)     // 48 total chunks
#define PFDIST  8             // prefetch distance in K-chunks

#if __has_builtin(__builtin_amdgcn_sched_barrier)
#define SCHED_FENCE() __builtin_amdgcn_sched_barrier(0)
#else
#define SCHED_FENCE() asm volatile("" ::: "memory")
#endif

typedef __attribute__((ext_vector_type(16))) _Float16 v16h;
typedef __attribute__((ext_vector_type(8)))  _Float16 v8h;
typedef __attribute__((ext_vector_type(8)))  float    v8f;

static __device__ __forceinline__ v16h cat8(v8h lo, v8h hi) {
    return __builtin_shufflevector(lo, hi, 0,1,2,3,4,5,6,7,8,9,10,11,12,13,14,15);
}

static __device__ __forceinline__ float sigmoidf_(float x) {
    return 1.0f / (1.0f + __expf(-x));
}

// WGP-scope prefetch (default scope == CU/WGP; pulls into all cache levels).
static __device__ __forceinline__ void pf_wgp(const void* p) {
    asm volatile("global_prefetch_b8 %0, off" :: "v"(p));
}

// register fragment for one K-chunk: A(16x32) + B(32x16) for 4 output column tiles
struct KFrag {
    v16h a;
    v16h b0, b1, b2, b3;
};

static __device__ __forceinline__ void load_frag(
        KFrag& f, int kk,
        const _Float16* __restrict__ xrow,   // rows for chunks < XCH
        const _Float16* __restrict__ hrow,   // rows for chunks >= XCH
        const _Float16* __restrict__ bc0, const _Float16* __restrict__ bc1,
        const _Float16* __restrict__ bc2, const _Float16* __restrict__ bc3,
        int hi8, int hi16) {
    const _Float16* ap = (kk < XCH) ? (xrow + kk * 32 + hi16)
                                    : (hrow + (kk * 32 - EDIM) + hi16);
    f.a = cat8(*(const v8h*)ap, *(const v8h*)(ap + 8));
    const int kb = kk * 32 + hi16;
    f.b0 = *(const v16h*)(bc0 + kb);
    f.b1 = *(const v16h*)(bc1 + kb);
    f.b2 = *(const v16h*)(bc2 + kb);
    f.b3 = *(const v16h*)(bc3 + kb);
}

static __device__ __forceinline__ void pf_chunk(
        int kk, int k1,
        const _Float16* __restrict__ xrow, const _Float16* __restrict__ hrow,
        const _Float16* __restrict__ bc0, const _Float16* __restrict__ bc1,
        const _Float16* __restrict__ bc2, const _Float16* __restrict__ bc3,
        int hi8, int hi16) {
    const int kp = kk + PFDIST;
    if (kp >= k1) return;    // wave-uniform guard
    const _Float16* ap = (kp < XCH) ? (xrow + kp * 32 + hi8)
                                    : (hrow + (kp * 32 - EDIM) + hi8);
    const int kb = kp * 32 + hi16;
    pf_wgp(ap);
    pf_wgp(bc0 + kb);
    pf_wgp(bc1 + kb);
    pf_wgp(bc2 + kb);
    pf_wgp(bc3 + kb);
}

static __device__ __forceinline__ void mm4(const KFrag& f, v8f acc[4]) {
    acc[0] = __builtin_amdgcn_wmma_f32_16x16x32_f16(false, f.a, false, f.b0, (short)0, acc[0], false, false);
    acc[1] = __builtin_amdgcn_wmma_f32_16x16x32_f16(false, f.a, false, f.b1, (short)0, acc[1], false, false);
    acc[2] = __builtin_amdgcn_wmma_f32_16x16x32_f16(false, f.a, false, f.b2, (short)0, acc[2], false, false);
    acc[3] = __builtin_amdgcn_wmma_f32_16x16x32_f16(false, f.a, false, f.b3, (short)0, acc[3], false, false);
}

// Fenced 2-stage software-pipelined K loop + WGP-scope prefetch stream.
// Requires (k1-k0) even, >= 2.
static __device__ __forceinline__ void gemm_k_pipe(
        v8f acc[4], int k0, int k1,
        const _Float16* __restrict__ xrow, const _Float16* __restrict__ hrow,
        const _Float16* __restrict__ bc0, const _Float16* __restrict__ bc1,
        const _Float16* __restrict__ bc2, const _Float16* __restrict__ bc3,
        int hi8, int hi16) {
    KFrag f0, f1;
    load_frag(f0, k0, xrow, hrow, bc0, bc1, bc2, bc3, hi8, hi16);
    pf_chunk(k0, k1, xrow, hrow, bc0, bc1, bc2, bc3, hi8, hi16);
    SCHED_FENCE();
    int kk = k0;
    for (; kk + 2 < k1; kk += 2) {
        load_frag(f1, kk + 1, xrow, hrow, bc0, bc1, bc2, bc3, hi8, hi16);
        pf_chunk(kk + 1, k1, xrow, hrow, bc0, bc1, bc2, bc3, hi8, hi16);
        SCHED_FENCE();
        mm4(f0, acc);
        SCHED_FENCE();
        load_frag(f0, kk + 2, xrow, hrow, bc0, bc1, bc2, bc3, hi8, hi16);
        pf_chunk(kk + 2, k1, xrow, hrow, bc0, bc1, bc2, bc3, hi8, hi16);
        SCHED_FENCE();
        mm4(f1, acc);
        SCHED_FENCE();
    }
    // drain: f0 holds chunk k1-2
    load_frag(f1, k1 - 1, xrow, hrow, bc0, bc1, bc2, bc3, hi8, hi16);
    SCHED_FENCE();
    mm4(f0, acc);
    mm4(f1, acc);
}

// ---------------- prep kernels ----------------

__global__ void k_cvt_x(const float* __restrict__ x, _Float16* __restrict__ xf, int n) {
    int s = gridDim.x * blockDim.x;
    for (int i = blockIdx.x * blockDim.x + threadIdx.x; i < n; i += s)
        xf[i] = (_Float16)x[i];
}

// fused weight, column-major: wc[n][k] = W_g[k][j], n = g*H + j  (f16)
__global__ void k_pack_w(const float* __restrict__ Wi, const float* __restrict__ Wo,
                         const float* __restrict__ Wf, const float* __restrict__ Wm,
                         _Float16* __restrict__ wc) {
    const int total = NG * KF;
    int s = gridDim.x * blockDim.x;
    for (int i = blockIdx.x * blockDim.x + threadIdx.x; i < total; i += s) {
        int n = i / KF;
        int k = i - n * KF;
        int g = n >> 10;
        int j = n & (HDIM - 1);
        const float* Wg = (g == 0) ? Wi : (g == 1) ? Wo : (g == 2) ? Wf : Wm;
        wc[i] = (_Float16)Wg[(size_t)k * HDIM + j];
    }
}

__global__ void k_pack_bias(const float* __restrict__ bi, const float* __restrict__ bo,
                            const float* __restrict__ bf, const float* __restrict__ bm,
                            float* __restrict__ bias) {
    int i = blockIdx.x * blockDim.x + threadIdx.x;
    if (i < NG) {
        int g = i >> 10;
        int j = i & (HDIM - 1);
        const float* bg = (g == 0) ? bi : (g == 1) ? bo : (g == 2) ? bf : bm;
        bias[i] = bg[j];
    }
}

__global__ void k_init_state(_Float16* __restrict__ h0, _Float16* __restrict__ h1,
                             float* __restrict__ c) {
    const int nh = BATCH * HDIM;
    int s = gridDim.x * blockDim.x;
    for (int i = blockIdx.x * blockDim.x + threadIdx.x; i < nh; i += s) {
        h0[i] = (_Float16)0.0f;
        h1[i] = (_Float16)0.0f;
        c[i]  = 0.0f;
    }
}

// ---------------- parallel input pre-GEMM: zx[t*B+m][n] = x @ Wx ----------------
__global__ void k_gemm_x(const _Float16* __restrict__ xf, const _Float16* __restrict__ wc,
                         _Float16* __restrict__ zx) {
    const int lane = threadIdx.x & 31;
    const int col  = lane & 15;
    const int hi8  = (lane >> 4) * 8;    // C-row base / A prefetch offset
    const int hi16 = (lane >> 4) * 16;   // B k-offset
    const int wid  = (blockIdx.x * blockDim.x + threadIdx.x) >> 5;
    const int nw   = (gridDim.x * blockDim.x) >> 5;
    const int ntiles = (T_STEPS * BATCH / 16) * (NG / 64);   // 2048 * 64

    for (int tile = wid; tile < ntiles; tile += nw) {
        const int mt = tile >> 6;
        const int n0 = (tile & 63) * 64;
        const _Float16* arow = xf + (size_t)(mt * 16 + col) * EDIM;
        const _Float16* bc0 = wc + (size_t)(n0 +  0 + col) * KF;
        const _Float16* bc1 = wc + (size_t)(n0 + 16 + col) * KF;
        const _Float16* bc2 = wc + (size_t)(n0 + 32 + col) * KF;
        const _Float16* bc3 = wc + (size_t)(n0 + 48 + col) * KF;

        v8f acc[4] = {};
        gemm_k_pipe(acc, 0, XCH, arow, arow, bc0, bc1, bc2, bc3, hi8, hi16);

#pragma unroll
        for (int q = 0; q < 4; ++q) {
            const int n = n0 + q * 16 + col;
#pragma unroll
            for (int r = 0; r < 8; ++r)
                zx[(size_t)(mt * 16 + hi8 + r) * NG + n] = (_Float16)acc[q][r];
        }
    }
}

// ---------------- serial recurrent step ----------------
// 64 blocks x 8 waves. Wave pair = (same 16x16x4gate tile, K split in half).
__global__ void __launch_bounds__(256)
k_lstm_step(const _Float16* __restrict__ xf,
            const _Float16* __restrict__ hin,
            _Float16* __restrict__ hout,
            float* __restrict__ cbuf,
            const _Float16* __restrict__ wc,
            const float* __restrict__ bias,
            const _Float16* __restrict__ zx,   // null => fused K=1536
            float* __restrict__ out,
            int t, int use_zx) {
    __shared__ float red[4][4][8 * 32];   // [jsub][gate][lane*8 + r] = 16 KB

    const int lane  = threadIdx.x & 31;
    const int col   = lane & 15;
    const int hi8   = (lane >> 4) * 8;
    const int hi16  = (lane >> 4) * 16;
    const int wave  = threadIdx.x >> 5;   // 0..7
    const int jsub  = wave >> 1;          // 0..3
    const int khalf = wave & 1;           // K split
    const int mt    = blockIdx.x >> 4;                    // 0..3
    const int j0    = ((blockIdx.x & 15) * 4 + jsub) * 16;
    const int m     = mt * 16 + col;      // A load row (batch index)

    const _Float16* xrow = xf + ((size_t)t * BATCH + m) * EDIM;
    const _Float16* hrow = hin + (size_t)m * HDIM;
    const _Float16* bc0 = wc + (size_t)(0 * HDIM + j0 + col) * KF;
    const _Float16* bc1 = wc + (size_t)(1 * HDIM + j0 + col) * KF;
    const _Float16* bc2 = wc + (size_t)(2 * HDIM + j0 + col) * KF;
    const _Float16* bc3 = wc + (size_t)(3 * HDIM + j0 + col) * KF;

    const int k_begin = use_zx ? XCH : 0;
    const int half    = (ALLCH - k_begin) >> 1;         // 16 or 24 (even)
    const int k0 = k_begin + khalf * half;
    const int k1 = k0 + half;

    v8f acc[4] = {};
    gemm_k_pipe(acc, k0, k1, xrow, hrow, bc0, bc1, bc2, bc3, hi8, hi16);

    // cross-wave split-K reduction via LDS
    if (khalf == 1) {
#pragma unroll
        for (int g = 0; g < 4; ++g)
            *(v8f*)&red[jsub][g][lane * 8] = acc[g];
    }
    __syncthreads();
    if (khalf == 1) return;

#pragma unroll
    for (int g = 0; g < 4; ++g)
        acc[g] += *(const v8f*)&red[jsub][g][lane * 8];

    // epilogue: gates + cell update (fp32), write h (f16 + fp32 out)
    const int j = j0 + col;
    float bg[4];
#pragma unroll
    for (int g = 0; g < 4; ++g) bg[g] = bias[g * HDIM + j];

#pragma unroll
    for (int r = 0; r < 8; ++r) {
        const int row = mt * 16 + hi8 + r;
        float z0 = acc[0][r] + bg[0];
        float z1 = acc[1][r] + bg[1];
        float z2 = acc[2][r] + bg[2];
        float z3 = acc[3][r] + bg[3];
        if (use_zx) {
            const _Float16* zr = zx + ((size_t)t * BATCH + row) * NG + j;
            z0 += (float)zr[0 * HDIM];
            z1 += (float)zr[1 * HDIM];
            z2 += (float)zr[2 * HDIM];
            z3 += (float)zr[3 * HDIM];
        }
        const float ig = sigmoidf_(z0);
        const float og = sigmoidf_(z1);
        const float fg = sigmoidf_(z2);
        const float mg = tanhf(z3);
        const size_t cidx = (size_t)row * HDIM + j;
        const float cn = fg * cbuf[cidx] + ig * mg;
        cbuf[cidx] = cn;
        const float hn = og * tanhf(cn);
        hout[cidx] = (_Float16)hn;
        out[(size_t)row * ((size_t)T_STEPS * HDIM) + (size_t)t * HDIM + j] = hn;
    }
}

// ---------------- host ----------------

extern "C" void kernel_launch(void* const* d_in, const int* in_sizes, int n_in,
                              void* d_out, int out_size, void* d_ws, size_t ws_size,
                              hipStream_t stream) {
    (void)in_sizes; (void)n_in; (void)out_size;
    const float* emb = (const float*)d_in[0];
    const float* Wi  = (const float*)d_in[1];
    const float* bi  = (const float*)d_in[2];
    const float* Wo  = (const float*)d_in[3];
    const float* bo  = (const float*)d_in[4];
    const float* Wf  = (const float*)d_in[5];
    const float* bf  = (const float*)d_in[6];
    const float* Wm  = (const float*)d_in[7];
    const float* bm  = (const float*)d_in[8];

    char* ws = (char*)d_ws;
    size_t off = 0;
    auto take = [&](size_t bytes) -> char* {
        char* p = ws + off;
        off = (off + bytes + 255) & ~(size_t)255;
        return p;
    };
    _Float16* xf   = (_Float16*)take((size_t)T_STEPS * BATCH * EDIM * sizeof(_Float16));
    _Float16* wc   = (_Float16*)take((size_t)NG * KF * sizeof(_Float16));
    float*    bias = (float*)   take((size_t)NG * sizeof(float));
    _Float16* h0   = (_Float16*)take((size_t)BATCH * HDIM * sizeof(_Float16));
    _Float16* h1   = (_Float16*)take((size_t)BATCH * HDIM * sizeof(_Float16));
    float*    cbuf = (float*)   take((size_t)BATCH * HDIM * sizeof(float));
    _Float16* zx   = (_Float16*)take((size_t)T_STEPS * BATCH * NG * sizeof(_Float16));
    const size_t need_zx = off;

    const int use_zx = (ws_size >= need_zx) ? 1 : 0;

    k_cvt_x     <<<4096, 256, 0, stream>>>(emb, xf, T_STEPS * BATCH * EDIM);
    k_pack_w    <<<4096, 256, 0, stream>>>(Wi, Wo, Wf, Wm, wc);
    k_pack_bias <<<(NG + 255) / 256, 256, 0, stream>>>(bi, bo, bf, bm, bias);
    k_init_state<<<256, 256, 0, stream>>>(h0, h1, cbuf);

    if (use_zx)
        k_gemm_x<<<2048, 256, 0, stream>>>(xf, wc, zx);

    float* out = (float*)d_out;
    for (int t = 0; t < T_STEPS; ++t) {
        const _Float16* hin = (t & 1) ? h1 : h0;
        _Float16*      hout = (t & 1) ? h0 : h1;
        k_lstm_step<<<64, 256, 0, stream>>>(xf, hin, hout, cbuf, wc, bias,
                                            use_zx ? zx : (const _Float16*)nullptr,
                                            out, t, use_zx);
    }
}
